// FusedMoE_64115271794786
// MI455X (gfx1250) — matrix operand order repaired and checked by reading, not compiled, not run on hardware
//
#include <hip/hip_runtime.h>

// ---------------------------------------------------------------------------
// Fused MoE for gfx1250 (MI455X), bf16 WMMA + async global->LDS path.
//   d_ws layout (requires ws_size >= ~49 MB):
//     [0      .. 64)        int   cnt[16]            (zeroed per call)
//     [1024   .. 33792)     float pairGate[8192]     (pid = tok*2+choice)
//     [65536  .. 327680)    int   list[16][4096]     (pid lists per expert)
//     [1MB    .. 1MB+47MB)  bf16  act[8192][2880]    (SiLU(gate)*up)
// ---------------------------------------------------------------------------

#define NEXP   16
#define HIDDEN 2880
#define INTER  2880
#define NTOK   4096
#define LDH    40            // LDS row stride in halves: 32 data + 8 pad (80B)
#define NK     (HIDDEN / 32) // 90 K-steps

typedef __attribute__((ext_vector_type(4)))  float  fx4;
typedef __attribute__((ext_vector_type(8)))  float  v8f;
typedef __attribute__((ext_vector_type(8)))  __bf16 bh8;
typedef __attribute__((ext_vector_type(16))) __bf16 bh16;

__device__ __forceinline__ bh8 cvt8(fx4 a, fx4 b) {
    bh8 r;
    r[0]=(__bf16)a[0]; r[1]=(__bf16)a[1]; r[2]=(__bf16)a[2]; r[3]=(__bf16)a[3];
    r[4]=(__bf16)b[0]; r[5]=(__bf16)b[1]; r[6]=(__bf16)b[2]; r[7]=(__bf16)b[3];
    return r;
}

__device__ __forceinline__ bh8 zero8() {
    bh8 z;
#pragma unroll
    for (int i = 0; i < 8; ++i) z[i] = (__bf16)0.0f;
    return z;
}

__device__ __forceinline__ v8f zerov8f() {
    v8f z;
#pragma unroll
    for (int i = 0; i < 8; ++i) z[i] = 0.0f;
    return z;
}

// Two ds_load_b128 from LDS, concatenated into a 16-element bf16 fragment.
__device__ __forceinline__ bh16 ldfrag(const __bf16* p, int off2) {
    bh8 lo = *reinterpret_cast<const bh8*>(p);
    bh8 hi = *reinterpret_cast<const bh8*>(p + off2);
    return __builtin_shufflevector(lo, hi, 0,1,2,3,4,5,6,7,8,9,10,11,12,13,14,15);
}

struct F16r { fx4 a, b, c, d; };
__device__ __forceinline__ F16r ldg16(const float* s) {
    F16r r;
    r.a = *reinterpret_cast<const fx4*>(s);
    r.b = *reinterpret_cast<const fx4*>(s + 4);
    r.c = *reinterpret_cast<const fx4*>(s + 8);
    r.d = *reinterpret_cast<const fx4*>(s + 12);
    return r;
}
__device__ __forceinline__ void stl16(__bf16* dst, F16r v) {
    *reinterpret_cast<bh8*>(dst)     = cvt8(v.a, v.b);
    *reinterpret_cast<bh8*>(dst + 8) = cvt8(v.c, v.d);
}

// Async DMA 32B global->LDS (two GLOBAL_LOAD_ASYNC_TO_LDS_B128).
// VDST operand = LDS byte offset = low 32 bits of the flat shared address
// (ISA 10.2: LDS aperture maps addr[31:0] directly to LDS).
__device__ __forceinline__ void async_cp32B(__bf16* dst, const __bf16* src) {
    unsigned off = (unsigned)(unsigned long long)(const void*)dst;
    asm volatile("global_load_async_to_lds_b128 %0, %1, off"
                 :: "v"(off), "v"(src) : "memory");
    asm volatile("global_load_async_to_lds_b128 %0, %1, off"
                 :: "v"(off + 16u), "v"(src + 8) : "memory");
}
__device__ __forceinline__ void wait_async0() {
    asm volatile("s_wait_asynccnt 0x0" ::: "memory");
}

// ---------------------------------------------------------------------------
// Kernel 1: routing. top-2 of softmax == top-2 of logits; renormalized gates
// reduce to a logistic of the logit gap (softmax denominator cancels).
// ---------------------------------------------------------------------------
__global__ __launch_bounds__(256) void moe_route(const float* __restrict__ rl,
                                                 int* __restrict__ cnt,
                                                 int* __restrict__ list,
                                                 float* __restrict__ gate) {
    int t = blockIdx.x * 256 + threadIdx.x;
    if (t >= NTOK) return;
    const float* p = rl + (size_t)t * NEXP;
    float l[NEXP];
#pragma unroll
    for (int i = 0; i < NEXP; ++i) l[i] = p[i];
    int i0 = 0; float b0 = l[0];
#pragma unroll
    for (int i = 1; i < NEXP; ++i) if (l[i] > b0) { b0 = l[i]; i0 = i; }
    int i1 = -1; float b1 = -3.4e38f;
#pragma unroll
    for (int i = 0; i < NEXP; ++i) if (i != i0 && l[i] > b1) { b1 = l[i]; i1 = i; }
    float g0 = 1.0f / (1.0f + __expf(b1 - b0));
    float g1 = 1.0f - g0;
    int s0 = atomicAdd(&cnt[i0], 1);
    list[i0 * NTOK + s0] = t * 2;
    gate[t * 2] = g0;
    int s1 = atomicAdd(&cnt[i1], 1);
    list[i1 * NTOK + s1] = t * 2 + 1;
    gate[t * 2 + 1] = g1;
}

// ---------------------------------------------------------------------------
// Kernel 2: act[pid] = SiLU(X@W13g^T + bg) * (X@W13u^T + bu), bf16 output.
// Block 128(M) x 64(N), 8 waves (2M x 4N), wave tile 64x16, K-step 32.
// Double-buffered LDS, software pipelined: global loads for k+1 issue before
// the k WMMAs; convert+ds_store after; one barrier per K-step.
// ---------------------------------------------------------------------------
struct __align__(16) Smem1 {
    __bf16 As[2 * 128 * LDH];   // token rows   [buf][128][32]
    __bf16 Bs[2 * 128 * LDH];   // gate(0..63)+up(64..127)
    int    Pids[128];
};

__global__ __launch_bounds__(256) void moe_gemm1(const float* __restrict__ X,
                                                 const float* __restrict__ W13,
                                                 const float* __restrict__ B13,
                                                 const int* __restrict__ cnt,
                                                 const int* __restrict__ list,
                                                 __bf16* __restrict__ act) {
    const int e = blockIdx.z;
    const int count = cnt[e];
    const int m0 = blockIdx.y * 128;
    if (m0 >= count) return;
    const int n0 = blockIdx.x * 64;

    const int tid = threadIdx.x;
    const int lane = tid & 31;
    const int w  = tid >> 5;
    const int wm = w & 1;        // M half
    const int wn = w >> 1;       // N 16-col slice

    __shared__ Smem1 sm;

    const int* lst = list + e * NTOK;
    if (tid < 128) {
        int m = m0 + tid;
        sm.Pids[tid] = (m < count) ? lst[m] : -1;
    }
    __syncthreads();

    const int frow = tid >> 1;                 // fill row 0..127
    const int fcb  = (tid & 1) << 4;           // col base 0 / 16
    const int apid = sm.Pids[frow];
    const float* W13e = W13 + (size_t)e * (2 * INTER) * HIDDEN;
    const size_t grow = (frow < 64) ? (size_t)(n0 + frow)
                                    : (size_t)(INTER + n0 + (frow - 64));
    const float* asrc = (apid >= 0) ? X + (size_t)(apid >> 1) * HIDDEN + fcb : X;
    const float* bsrc = W13e + grow * HIDDEN + fcb;

    // Out-of-range rows are k-invariant: zero them once in both buffers.
    if (apid < 0) {
#pragma unroll
        for (int b = 0; b < 2; ++b) {
            __bf16* d = &sm.As[b * 128 * LDH + frow * LDH + fcb];
            *reinterpret_cast<bh8*>(d)     = zero8();
            *reinterpret_cast<bh8*>(d + 8) = zero8();
        }
    }
    // Prologue: stage k=0 into buffer 0.
    if (apid >= 0) stl16(&sm.As[frow * LDH + fcb], ldg16(asrc));
    stl16(&sm.Bs[frow * LDH + fcb], ldg16(bsrc));
    __syncthreads();

    v8f cg[4], cu[4];
#pragma unroll
    for (int i = 0; i < 4; ++i) { cg[i] = zerov8f(); cu[i] = zerov8f(); }

    const int l15 = lane & 15;
    const int ah  = (lane & 16) >> 1;   // A chunks at {ah, ah+16} halves
    const int bh_ = (lane & 16) >> 1;   // B chunks at {bh, bh+16} halves

    for (int kk = 0; kk < NK; ++kk) {
        const bool more = (kk + 1 < NK);
        const int  k1 = (kk + 1) * 32;
        F16r ra, rb;
        if (more) {                               // global loads first (latency)
            if (apid >= 0) ra = ldg16(asrc + k1);
            rb = ldg16(bsrc + k1);
            __builtin_prefetch(bsrc + k1 + 32, 0, 1);   // k+2 weight row
        }
        {   // compute from buffer kk&1: 8 x v_wmma_f32_16x16x32_bf16 per wave
            const __bf16* Ab = &sm.As[(kk & 1) * 128 * LDH];
            const __bf16* Bb = &sm.Bs[(kk & 1) * 128 * LDH];
            bh16 bg = ldfrag(&Bb[(wn * 16 + l15) * LDH + bh_], 16);
            bh16 bu = ldfrag(&Bb[(64 + wn * 16 + l15) * LDH + bh_], 16);
#pragma unroll
            for (int i = 0; i < 4; ++i) {
                bh16 a = ldfrag(&Ab[(wm * 64 + i * 16 + l15) * LDH + ah], 16);
                cg[i] = __builtin_amdgcn_wmma_f32_16x16x32_bf16(
                            false, a, false, bg, (short)0, cg[i], false, false);
                cu[i] = __builtin_amdgcn_wmma_f32_16x16x32_bf16(
                            false, a, false, bu, (short)0, cu[i], false, false);
            }
        }
        if (more) {                               // convert + stage into buf^1
            const int nb = (kk + 1) & 1;
            if (apid >= 0) stl16(&sm.As[nb * 128 * LDH + frow * LDH + fcb], ra);
            stl16(&sm.Bs[nb * 128 * LDH + frow * LDH + fcb], rb);
        }
        __syncthreads();
    }

    // --- epilogue: bias + SiLU(gate)*up -> bf16 act[pid][n] ---
    const int n  = n0 + wn * 16 + l15;
    const float gb = B13[(size_t)e * (2 * INTER) + n];
    const float ub = B13[(size_t)e * (2 * INTER) + INTER + n];
    const int mb = wm * 64 + ((lane >> 4) << 3);   // C layout: M = j + 8*(lane/16)
#pragma unroll
    for (int i = 0; i < 4; ++i) {
#pragma unroll
        for (int j = 0; j < 8; ++j) {
            int ml = mb + i * 16 + j;
            int pid = sm.Pids[ml];
            if (pid >= 0) {
                float gv = cg[i][j] + gb;
                float uv = cu[i][j] + ub;
                float s = gv / (1.0f + __expf(-gv));
                act[(size_t)pid * INTER + n] = (__bf16)(s * uv);
            }
        }
    }
}

// ---------------------------------------------------------------------------
// Kernel 3: out[tok] += gate * (act@W2^T + b2). A-tile (bf16 act) is staged
// with GLOBAL_LOAD_ASYNC_TO_LDS_B128: tracked by ASYNCcnt (not DScnt), so the
// DMA runs concurrent with the WMMA ds_loads; s_wait_asynccnt 0 before the
// barrier. Scatter via global_atomic_add_f32 (2 addends/element).
// ---------------------------------------------------------------------------
struct __align__(16) Smem2 {
    __bf16 As[2 * 128 * LDH];
    __bf16 Bs[2 * 64 * LDH];
    int    Pids[128];
    float  Gs[128];
};

__global__ __launch_bounds__(256) void moe_gemm2(const __bf16* __restrict__ act,
                                                 const float* __restrict__ W2,
                                                 const float* __restrict__ B2,
                                                 const int* __restrict__ cnt,
                                                 const int* __restrict__ list,
                                                 const float* __restrict__ gate,
                                                 float* __restrict__ out) {
    const int e = blockIdx.z;
    const int count = cnt[e];
    const int m0 = blockIdx.y * 128;
    if (m0 >= count) return;
    const int n0 = blockIdx.x * 64;

    const int tid = threadIdx.x;
    const int lane = tid & 31;
    const int w  = tid >> 5;
    const int wm = w & 1;
    const int wn = w >> 1;

    __shared__ Smem2 sm;

    const int* lst = list + e * NTOK;
    if (tid < 128) {
        int m = m0 + tid;
        int pid = (m < count) ? lst[m] : -1;
        sm.Pids[tid] = pid;
        sm.Gs[tid] = (pid >= 0) ? gate[pid] : 0.0f;
    }
    __syncthreads();

    const int arow = tid >> 1;              // A fill: 128 rows x 32 halves
    const int acb  = (tid & 1) << 4;
    const int brow = tid >> 2;              // B fill: 64 rows x 32 f32
    const int bcb  = (tid & 3) << 3;
    const int apid = sm.Pids[arow];
    const __bf16* asrc = (apid >= 0) ? act + (size_t)apid * INTER + acb : act;
    const float*  bsrc = W2 + (size_t)e * HIDDEN * INTER
                            + (size_t)(n0 + brow) * INTER + bcb;

    if (apid < 0) {
#pragma unroll
        for (int b = 0; b < 2; ++b) {
            __bf16* d = &sm.As[b * 128 * LDH + arow * LDH + acb];
            *reinterpret_cast<bh8*>(d)     = zero8();
            *reinterpret_cast<bh8*>(d + 8) = zero8();
        }
    }
    // Prologue: k=0 into buffer 0 (A via async DMA, B via convert path).
    if (apid >= 0) async_cp32B(&sm.As[arow * LDH + acb], asrc);
    {
        fx4 x0 = *reinterpret_cast<const fx4*>(bsrc);
        fx4 x1 = *reinterpret_cast<const fx4*>(bsrc + 4);
        *reinterpret_cast<bh8*>(&sm.Bs[brow * LDH + bcb]) = cvt8(x0, x1);
    }
    wait_async0();
    __syncthreads();

    v8f c[4];
#pragma unroll
    for (int i = 0; i < 4; ++i) c[i] = zerov8f();

    const int l15 = lane & 15;
    const int ah  = (lane & 16) >> 1;
    const int bh_ = (lane & 16) >> 1;

    for (int kk = 0; kk < NK; ++kk) {
        const bool more = (kk + 1 < NK);
        const int  k1 = (kk + 1) * 32;
        const int  nb = (kk + 1) & 1;
        fx4 rb0, rb1;
        if (more) {
            if (apid >= 0)                       // DMA next A tile (ASYNCcnt)
                async_cp32B(&sm.As[nb * 128 * LDH + arow * LDH + acb], asrc + k1);
            rb0 = *reinterpret_cast<const fx4*>(bsrc + k1);
            rb1 = *reinterpret_cast<const fx4*>(bsrc + k1 + 4);
            __builtin_prefetch(bsrc + k1 + 32, 0, 1);
        }
        {   // compute from buffer kk&1: 4 x v_wmma_f32_16x16x32_bf16 per wave
            const __bf16* Ab = &sm.As[(kk & 1) * 128 * LDH];
            const __bf16* Bb = &sm.Bs[(kk & 1) * 64 * LDH];
            bh16 b = ldfrag(&Bb[(wn * 16 + l15) * LDH + bh_], 16);
#pragma unroll
            for (int i = 0; i < 4; ++i) {
                bh16 a = ldfrag(&Ab[(wm * 64 + i * 16 + l15) * LDH + ah], 16);
                c[i] = __builtin_amdgcn_wmma_f32_16x16x32_bf16(
                           false, a, false, b, (short)0, c[i], false, false);
            }
        }
        if (more)
            *reinterpret_cast<bh8*>(&sm.Bs[nb * 64 * LDH + brow * LDH + bcb]) =
                cvt8(rb0, rb1);
        wait_async0();
        __syncthreads();
    }

    // --- epilogue: + bias, * gate, atomic scatter into out[token][n] ---
    const int n = n0 + wn * 16 + l15;
    const float bn = B2[(size_t)e * HIDDEN + n];
    const int mb = wm * 64 + ((lane >> 4) << 3);
#pragma unroll
    for (int i = 0; i < 4; ++i) {
#pragma unroll
        for (int j = 0; j < 8; ++j) {
            int ml = mb + i * 16 + j;
            int pid = sm.Pids[ml];
            if (pid >= 0) {
                float y = c[i][j] + bn;
                unsafeAtomicAdd(&out[(size_t)(pid >> 1) * HIDDEN + n], sm.Gs[ml] * y);
            }
        }
    }
}

// ---------------------------------------------------------------------------
extern "C" void kernel_launch(void* const* d_in, const int* in_sizes, int n_in,
                              void* d_out, int out_size, void* d_ws, size_t ws_size,
                              hipStream_t stream) {
    const float* X   = (const float*)d_in[0];  // hidden_states [4096][2880]
    const float* RL  = (const float*)d_in[1];  // router_logits [4096][16]
    const float* W13 = (const float*)d_in[2];  // [16][5760][2880]
    const float* W2  = (const float*)d_in[3];  // [16][2880][2880]
    const float* B13 = (const float*)d_in[4];  // [16][5760]
    const float* B2  = (const float*)d_in[5];  // [16][2880]
    float* out = (float*)d_out;

    char* ws = (char*)d_ws;
    int*    cnt  = (int*)(ws + 0);
    float*  gate = (float*)(ws + 1024);
    int*    list = (int*)(ws + 65536);
    __bf16* act  = (__bf16*)(ws + (1u << 20));

    hipMemsetAsync(cnt, 0, 64, stream);
    hipMemsetAsync(out, 0, (size_t)out_size * sizeof(float), stream);

    moe_route<<<NTOK / 256, 256, 0, stream>>>(RL, cnt, list, gate);

    dim3 g1(INTER / 64, NTOK / 128, NEXP);   // 45 x 32 x 16
    moe_gemm1<<<g1, 256, 0, stream>>>(X, W13, B13, cnt, list, act);

    dim3 g2(HIDDEN / 64, NTOK / 128, NEXP);  // 45 x 32 x 16
    moe_gemm2<<<g2, 256, 0, stream>>>(act, W2, B2, cnt, list, gate, out);
}
